// Coma_83863531422195
// MI455X (gfx1250) — matrix operand order repaired and checked by reading, not compiled, not run on hardware
//
#include <hip/hip_runtime.h>

// CoMA mesh autoencoder for MI455X (gfx1250, wave32, WMMA).
//
// Activations feature-major [C, N] (C = B*F, node innermost). Dense Chebyshev
// propagation computed transposed:  Y[c,m] = sum_n X[c,n] * adj[m,n]
// -> both WMMA operands read CONTIGUOUS memory (b128 loads only).
// The GEMM kernel is software-pipelined: the next k-step's 20 b128 loads are
// in flight while the current step converts f32->f16 and issues 4 WMMAs, so
// waits need only loadcnt<=20 (full k-step of latency hiding).
// adj1 (105 MB f32) stays L2-resident (192 MB L2).
// Workspace: 3 * (512*20480) floats + small (~126 MB).

#define B_    32
#define N0_   20480
#define N1_   5120
#define N2_   1280
#define N3_   320
#define N4_   80
#define KCH   6
#define ZDIM  64
#define EDGES (N0_ * 6)

typedef _Float16 v16h __attribute__((ext_vector_type(16)));
typedef float    v8f  __attribute__((ext_vector_type(8)));

// ---------------- elementwise helpers ----------------

__global__ void k_zero(float* __restrict__ p, int n) {
  int t = blockIdx.x * blockDim.x + threadIdx.x;
  if (t < n) p[t] = 0.0f;
}

__global__ void k_neg(float* __restrict__ p, int n) {
  int t = blockIdx.x * blockDim.x + threadIdx.x;
  if (t < n) p[t] = -p[t];
}

// x [B*N0, 3] batch-major  ->  o [(b*3+f), n] feature-major
__global__ void k_transpose_in(const float* __restrict__ x, float* __restrict__ o) {
  int t = blockIdx.x * blockDim.x + threadIdx.x;
  if (t >= B_ * 3 * N0_) return;
  int n = t % N0_;
  int c = t / N0_;
  int f = c % 3;
  int b = c / 3;
  o[t] = x[((size_t)b * N0_ + n) * 3 + f];
}

// ---------------- sparse level-0 propagation (scatter-add) ----------------
template <int F>
__global__ void k_scatter(float* __restrict__ out, const float* __restrict__ in,
                          const int* __restrict__ src, const int* __restrict__ dst,
                          const float* __restrict__ norm, float scale) {
  int t = blockIdx.x * blockDim.x + threadIdx.x;
  if (t >= EDGES * B_) return;
  int e = t % EDGES;        // e fastest: coalesced norm/src/dst reads
  int b = t / EDGES;
  float w = scale * norm[e];
  int s = src[e], d = dst[e];
  const float* xi = in + (size_t)b * F * N0_;
  float* yo = out + (size_t)b * F * N0_;
#pragma unroll
  for (int f = 0; f < F; ++f)
    atomicAdd(yo + (size_t)f * N0_ + d, w * xi[(size_t)f * N0_ + s]);
}

// ---------------- per-node weight transform ----------------
// out[(b,o), n] (+)= sum_f in[(b,f), n] * W[f,O]  (+bias, relu on final k)
// remap != 0: write batch-major d_out[(b*Nn + n)*O + o] instead.
template <int F, int O>
__global__ void k_wt(float* out, const float* __restrict__ in,
                     const float* __restrict__ W, const float* __restrict__ bias,
                     int Nn, int accum, int relu, int remap) {
  int t = blockIdx.x * blockDim.x + threadIdx.x;
  if (t >= B_ * Nn) return;
  int n = t % Nn;
  int b = t / Nn;
  const float* xi = in + (size_t)b * F * Nn + n;
  float xr[F];
#pragma unroll
  for (int f = 0; f < F; ++f) xr[f] = xi[(size_t)f * Nn];
  if (remap) {
    float* yo = out + ((size_t)b * Nn + n) * O;
#pragma unroll
    for (int o = 0; o < O; ++o) {
      float acc = accum ? yo[o] : 0.0f;
#pragma unroll
      for (int f = 0; f < F; ++f) acc = fmaf(xr[f], W[f * O + o], acc);
      if (bias) acc += bias[o];
      if (relu) acc = acc > 0.0f ? acc : 0.0f;
      yo[o] = acc;
    }
  } else {
    float* yo = out + (size_t)b * O * Nn + n;
#pragma unroll
    for (int o = 0; o < O; ++o) {
      float acc = accum ? yo[(size_t)o * Nn] : 0.0f;
#pragma unroll
      for (int f = 0; f < F; ++f) acc = fmaf(xr[f], W[f * O + o], acc);
      if (bias) acc += bias[o];
      if (relu) acc = acc > 0.0f ? acc : 0.0f;
      yo[(size_t)o * Nn] = acc;
    }
  }
}

// ---------------- pooling (<=3 nnz per row) ----------------
__global__ void k_pool(float* __restrict__ out, const float* __restrict__ in,
                       const int* __restrict__ idx, const float* __restrict__ w,
                       int Mout, int Nin, int F) {
  int t = blockIdx.x * blockDim.x + threadIdx.x;
  if (t >= B_ * F * Mout) return;
  int m = t % Mout;
  int c = t / Mout;
  const float* xi = in + (size_t)c * Nin;
  float acc = 0.0f;
#pragma unroll
  for (int j = 0; j < 3; ++j)
    acc = fmaf(w[m * 3 + j], xi[idx[m * 3 + j]], acc);
  out[t] = acc;
}

// ---------------- latent FC layers ----------------
__global__ void k_fc_enc(float* __restrict__ z, const float* __restrict__ h,
                         const float* __restrict__ w, const float* __restrict__ bia) {
  int t = blockIdx.x * blockDim.x + threadIdx.x;
  if (t >= B_ * ZDIM) return;
  int o = t % ZDIM;
  int b = t / ZDIM;
  float acc = bia[o];
  for (int n = 0; n < N4_; ++n)
#pragma unroll
    for (int f = 0; f < 32; ++f)
      acc = fmaf(h[((size_t)b * 32 + f) * N4_ + n], w[(size_t)o * (N4_ * 32) + n * 32 + f], acc);
  z[t] = acc > 0.0f ? acc : 0.0f;
}

__global__ void k_fc_dec(float* __restrict__ h, const float* __restrict__ z,
                         const float* __restrict__ w, const float* __restrict__ bia) {
  int t = blockIdx.x * blockDim.x + threadIdx.x;
  if (t >= B_ * 32 * N4_) return;
  int n = t % N4_;
  int c = t / N4_;
  int f = c % 32;
  int b = c / 32;
  int i = n * 32 + f;
  float acc = bia[i];
#pragma unroll
  for (int zz = 0; zz < ZDIM; ++zz)
    acc = fmaf(z[b * ZDIM + zz], w[(size_t)i * ZDIM + zz], acc);
  h[t] = acc > 0.0f ? acc : 0.0f;
}

// ---------------- WMMA dense propagation (software-pipelined) ----------------

__device__ __forceinline__ void load_k(const float* pa, const float* pb0,
                                       const float* pb1, const float* pb2,
                                       const float* pb3, int off, float4 R[20]) {
  const float4* q;
  q = (const float4*)(pa + off);
  R[0] = q[0]; R[1] = q[1]; R[2] = q[2]; R[3] = q[3];   // A: K {0..15}
  q = (const float4*)(pb0 + off);
  R[4] = q[0]; R[5] = q[1]; R[6] = q[2]; R[7] = q[3];
  q = (const float4*)(pb1 + off);
  R[8] = q[0]; R[9] = q[1]; R[10] = q[2]; R[11] = q[3];
  q = (const float4*)(pb2 + off);
  R[12] = q[0]; R[13] = q[1]; R[14] = q[2]; R[15] = q[3];
  q = (const float4*)(pb3 + off);
  R[16] = q[0]; R[17] = q[1]; R[18] = q[2]; R[19] = q[3];
}

__device__ __forceinline__ v16h cvt16(const float4& x0, const float4& x1,
                                      const float4& x2, const float4& x3) {
  v16h v;
  v[0]  = (_Float16)x0.x; v[1]  = (_Float16)x0.y; v[2]  = (_Float16)x0.z; v[3]  = (_Float16)x0.w;
  v[4]  = (_Float16)x1.x; v[5]  = (_Float16)x1.y; v[6]  = (_Float16)x1.z; v[7]  = (_Float16)x1.w;
  v[8]  = (_Float16)x2.x; v[9]  = (_Float16)x2.y; v[10] = (_Float16)x2.z; v[11] = (_Float16)x2.w;
  v[12] = (_Float16)x3.x; v[13] = (_Float16)x3.y; v[14] = (_Float16)x3.z; v[15] = (_Float16)x3.w;
  return v;
}

__device__ __forceinline__ void mma_step(const float4 R[20], v8f& c0, v8f& c1,
                                         v8f& c2, v8f& c3) {
  v16h a  = cvt16(R[0],  R[1],  R[2],  R[3]);
  v16h b0 = cvt16(R[4],  R[5],  R[6],  R[7]);
  v16h b1 = cvt16(R[8],  R[9],  R[10], R[11]);
  v16h b2 = cvt16(R[12], R[13], R[14], R[15]);
  v16h b3 = cvt16(R[16], R[17], R[18], R[19]);
  c0 = __builtin_amdgcn_wmma_f32_16x16x32_f16(false, a, false, b0, (short)0, c0, false, false);
  c1 = __builtin_amdgcn_wmma_f32_16x16x32_f16(false, a, false, b1, (short)0, c1, false, false);
  c2 = __builtin_amdgcn_wmma_f32_16x16x32_f16(false, a, false, b2, (short)0, c2, false, false);
  c3 = __builtin_amdgcn_wmma_f32_16x16x32_f16(false, a, false, b3, (short)0, c3, false, false);
}

// Y[c,m] = alpha * sum_n X[c,n]*adj[m,n] + beta * prev[c,m]
// X, Y, prev: [C, Nn]; adj: [Nn, Nn]. prev may be null or alias Y.
// Wave: 16(c) x 64(m) tile; block = 4 waves on 4 consecutive c-tiles.
// Nn % 64 == 0, C % 64 == 0, Nn/32 even and >= 10 for all levels.
__launch_bounds__(128)
__global__ void k_wmma_prop(float* out, const float* __restrict__ x,
                            const float* __restrict__ adj, const float* prev,
                            int Nn, int C, float alpha, float beta) {
  int lane = threadIdx.x & 31;
  int wv   = threadIdx.x >> 5;
  int rt   = blockIdx.x * 4 + wv;   // c row-tile
  int mt   = blockIdx.y;            // m col-tile (64 wide)
  int half = lane >> 4;
  int lm   = lane & 15;

  int c = rt * 16 + lm;
  const float* pa  = x + (size_t)c * Nn + half * 16;
  const float* pb0 = adj + (size_t)(mt * 64 +  0 + lm) * Nn + half * 16;
  const float* pb1 = adj + (size_t)(mt * 64 + 16 + lm) * Nn + half * 16;
  const float* pb2 = adj + (size_t)(mt * 64 + 32 + lm) * Nn + half * 16;
  const float* pb3 = adj + (size_t)(mt * 64 + 48 + lm) * Nn + half * 16;

  v8f acc0 = {}, acc1 = {}, acc2 = {}, acc3 = {};
  float4 R0[20], R1[20];

  load_k(pa, pb0, pb1, pb2, pb3, 0, R0);
  int steps = Nn >> 5;              // even, >= 10
  for (int i = 0; i < steps - 2; i += 2) {
    load_k(pa, pb0, pb1, pb2, pb3, 32, R1);   // prefetch step i+1
    mma_step(R0, acc0, acc1, acc2, acc3);     // consume step i
    pa += 64; pb0 += 64; pb1 += 64; pb2 += 64; pb3 += 64;
    load_k(pa, pb0, pb1, pb2, pb3, 0, R0);    // prefetch step i+2
    mma_step(R1, acc0, acc1, acc2, acc3);     // consume step i+1
  }
  load_k(pa, pb0, pb1, pb2, pb3, 32, R1);     // last step (no OOB prefetch)
  mma_step(R0, acc0, acc1, acc2, acc3);
  mma_step(R1, acc0, acc1, acc2, acc3);

  // ---- epilogue: D row = rt*16 + r + half*8, col = mt*64 + j*16 + lm
  int colb = mt * 64 + lm;
#pragma unroll
  for (int r = 0; r < 8; ++r) {
    int row = rt * 16 + r + half * 8;
    size_t o0 = (size_t)row * Nn + colb;
    float p0 = prev ? beta * prev[o0]      : 0.0f;
    float p1 = prev ? beta * prev[o0 + 16] : 0.0f;
    float p2 = prev ? beta * prev[o0 + 32] : 0.0f;
    float p3 = prev ? beta * prev[o0 + 48] : 0.0f;
    out[o0]      = fmaf(alpha, acc0[r], p0);
    out[o0 + 16] = fmaf(alpha, acc1[r], p1);
    out[o0 + 32] = fmaf(alpha, acc2[r], p2);
    out[o0 + 48] = fmaf(alpha, acc3[r], p3);
  }
}

// ---------------- host orchestration ----------------

static inline int cdiv(long long a, int b) { return (int)((a + b - 1) / b); }

extern "C" void kernel_launch(void* const* d_in, const int* in_sizes, int n_in,
                              void* d_out, int out_size, void* d_ws, size_t ws_size,
                              hipStream_t stream) {
  const float* x    = (const float*)d_in[0];
  const int*   ei   = (const int*)d_in[1];
  const int*   src  = ei;
  const int*   dst  = ei + EDGES;
  const float* norm = (const float*)d_in[2];
  const float* adj1 = (const float*)d_in[4];
  const float* adj2 = (const float*)d_in[5];
  const float* adj3 = (const float*)d_in[6];

  const int*   dn_idx[4] = {(const int*)d_in[7],  (const int*)d_in[11], (const int*)d_in[15], (const int*)d_in[19]};
  const float* dn_w[4]   = {(const float*)d_in[8], (const float*)d_in[12], (const float*)d_in[16], (const float*)d_in[20]};
  const int*   up_idx[4] = {(const int*)d_in[9],  (const int*)d_in[13], (const int*)d_in[17], (const int*)d_in[21]};
  const float* up_w[4]   = {(const float*)d_in[10], (const float*)d_in[14], (const float*)d_in[18], (const float*)d_in[22]};

  const float* Wenc[4] = {(const float*)d_in[23], (const float*)d_in[25], (const float*)d_in[27], (const float*)d_in[29]};
  const float* benc[4] = {(const float*)d_in[24], (const float*)d_in[26], (const float*)d_in[28], (const float*)d_in[30]};
  const float* Wdec[5] = {(const float*)d_in[31], (const float*)d_in[33], (const float*)d_in[35], (const float*)d_in[37], (const float*)d_in[39]};
  const float* bdec[4] = {(const float*)d_in[32], (const float*)d_in[34], (const float*)d_in[36], (const float*)d_in[38]};
  const float* enc_w = (const float*)d_in[40];
  const float* enc_b = (const float*)d_in[41];
  const float* dec_w = (const float*)d_in[42];
  const float* dec_b = (const float*)d_in[43];

  const size_t SB = (size_t)B_ * 16 * N0_;     // 10,485,760 floats
  float* A  = (float*)d_ws;
  float* Bb = A + SB;
  float* Cc = A + 2 * SB;
  float* zb = A + 3 * SB;                      // B_*ZDIM floats

  auto ZERO = [&](float* p, long long n) {
    k_zero<<<cdiv(n, 256), 256, 0, stream>>>(p, (int)n);
  };
  auto NEG = [&](float* p, long long n) {
    k_neg<<<cdiv(n, 256), 256, 0, stream>>>(p, (int)n);
  };
  auto SC = [&](float* o, const float* i, float s, int F) {
    long long n = (long long)EDGES * B_;
    if (F == 3)  k_scatter<3><<<cdiv(n, 256), 256, 0, stream>>>(o, i, src, dst, norm, s);
    else         k_scatter<16><<<cdiv(n, 256), 256, 0, stream>>>(o, i, src, dst, norm, s);
  };
  auto WT = [&](float* o, const float* i, const float* W, int k, const float* bias,
                int Nn, int F, int O, int accum, int relu, int remap) {
    const float* Wk = W + (size_t)k * F * O;
    dim3 g(cdiv((long long)B_ * Nn, 256)), b(256);
    if (F == 3  && O == 16)      k_wt<3, 16><<<g, b, 0, stream>>>(o, i, Wk, bias, Nn, accum, relu, remap);
    else if (F == 16 && O == 16) k_wt<16, 16><<<g, b, 0, stream>>>(o, i, Wk, bias, Nn, accum, relu, remap);
    else if (F == 16 && O == 32) k_wt<16, 32><<<g, b, 0, stream>>>(o, i, Wk, bias, Nn, accum, relu, remap);
    else if (F == 32 && O == 16) k_wt<32, 16><<<g, b, 0, stream>>>(o, i, Wk, bias, Nn, accum, relu, remap);
    else                         k_wt<16, 3><<<g, b, 0, stream>>>(o, i, Wk, bias, Nn, accum, relu, remap);
  };
  auto PROP = [&](float* o, const float* xin, const float* adj, const float* prev,
                  int Nn, int C, float al, float be) {
    dim3 g(C / 64, Nn / 64);
    k_wmma_prop<<<g, 128, 0, stream>>>(o, xin, adj, prev, Nn, C, al, be);
  };
  auto POOL = [&](float* o, const float* i, const int* idx, const float* w,
                  int Mout, int Nin, int F) {
    long long n = (long long)B_ * F * Mout;
    k_pool<<<cdiv(n, 256), 256, 0, stream>>>(o, i, idx, w, Mout, Nin, F);
  };

  // -- Chebyshev layer (sparse adjacency, level 0) ---------------------------
  auto CHEB_SPARSE = [&](float* t0, float* t1, float* outp, const float* W,
                         const float* bias, int F, int O, int relu, int remap) {
    long long nf = (long long)B_ * F * N0_;
    WT(outp, t0, W, 0, nullptr, N0_, F, O, 0, 0, remap);
    ZERO(t1, nf);
    SC(t1, t0, 1.0f, F);
    WT(outp, t1, W, 1, nullptr, N0_, F, O, 1, 0, remap);
    float* Ta = t0;  // Tx_{k-2}
    float* Tb = t1;  // Tx_{k-1}
    for (int k = 2; k < KCH; ++k) {
      NEG(Ta, nf);                 // Ta = -Tx_{k-2}
      SC(Ta, Tb, 2.0f, F);         // Ta += 2*prop(Tx_{k-1})  -> Tx_k
      int last = (k == KCH - 1);
      WT(outp, Ta, W, k, last ? bias : nullptr, N0_, F, O, 1, last ? relu : 0, remap);
      float* tmp = Ta; Ta = Tb; Tb = tmp;
    }
  };

  // -- Chebyshev layer (dense adjacency, WMMA) -------------------------------
  auto CHEB_DENSE = [&](float* t0, float* t1, float* outp, const float* adj, int Nn,
                        const float* W, const float* bias, int F, int O, int relu) {
    int C = B_ * F;
    WT(outp, t0, W, 0, nullptr, Nn, F, O, 0, 0, 0);
    PROP(t1, t0, adj, nullptr, Nn, C, 1.0f, 0.0f);          // Tx1 = adj @ Tx0
    WT(outp, t1, W, 1, nullptr, Nn, F, O, 1, 0, 0);
    float* Ta = t0;
    float* Tb = t1;
    for (int k = 2; k < KCH; ++k) {
      PROP(Ta, Tb, adj, Ta, Nn, C, 2.0f, -1.0f);            // Tx_k = 2 adj@Tx_{k-1} - Tx_{k-2}
      int last = (k == KCH - 1);
      WT(outp, Ta, W, k, last ? bias : nullptr, Nn, F, O, 1, last ? relu : 0, 0);
      float* tmp = Ta; Ta = Tb; Tb = tmp;
    }
  };

  // ====================== forward pass ======================
  k_transpose_in<<<cdiv((long long)B_ * 3 * N0_, 256), 256, 0, stream>>>(x, A);
  CHEB_SPARSE(A, Bb, Cc, Wenc[0], benc[0], 3, 16, 1, 0);
  POOL(A, Cc, dn_idx[0], dn_w[0], N1_, N0_, 16);
  CHEB_DENSE(A, Bb, Cc, adj1, N1_, Wenc[1], benc[1], 16, 16, 1);
  POOL(A, Cc, dn_idx[1], dn_w[1], N2_, N1_, 16);
  CHEB_DENSE(A, Bb, Cc, adj2, N2_, Wenc[2], benc[2], 16, 16, 1);
  POOL(A, Cc, dn_idx[2], dn_w[2], N3_, N2_, 16);
  CHEB_DENSE(A, Bb, Cc, adj3, N3_, Wenc[3], benc[3], 16, 32, 1);
  POOL(A, Cc, dn_idx[3], dn_w[3], N4_, N3_, 32);

  k_fc_enc<<<cdiv(B_ * ZDIM, 256), 256, 0, stream>>>(zb, A, enc_w, enc_b);
  k_fc_dec<<<cdiv(B_ * 32 * N4_, 256), 256, 0, stream>>>(A, zb, dec_w, dec_b);

  // decoder
  POOL(Bb, A, up_idx[3], up_w[3], N3_, N4_, 32);
  CHEB_DENSE(Bb, A, Cc, adj3, N3_, Wdec[0], bdec[0], 32, 16, 1);
  POOL(A, Cc, up_idx[2], up_w[2], N2_, N3_, 16);
  CHEB_DENSE(A, Bb, Cc, adj2, N2_, Wdec[1], bdec[1], 16, 16, 1);
  POOL(A, Cc, up_idx[1], up_w[1], N1_, N2_, 16);
  CHEB_DENSE(A, Bb, Cc, adj1, N1_, Wdec[2], bdec[2], 16, 16, 1);
  POOL(A, Cc, up_idx[0], up_w[0], N0_, N1_, 16);
  CHEB_SPARSE(A, Bb, Cc, Wdec[3], bdec[3], 16, 16, 1, 0);
  // final layer: accumulate directly into d_out batch-major, no bias/relu
  CHEB_SPARSE(Cc, A, (float*)d_out, Wdec[4], nullptr, 16, 3, 0, 1);

  (void)in_sizes; (void)n_in; (void)out_size; (void)ws_size;
}
